// SupervisedCoordContrastiveLoss_6279242187473
// MI455X (gfx1250) — matrix operand order repaired and checked by reading, not compiled, not run on hardware
//
#include <hip/hip_runtime.h>
#include <hip/hip_bf16.h>
#include <math.h>

typedef __bf16 bf16_t;
typedef bf16_t v16bf __attribute__((ext_vector_type(16)));
typedef float  v8f   __attribute__((ext_vector_type(8)));
typedef unsigned int u32x4 __attribute__((ext_vector_type(4)));
typedef unsigned int v4u __attribute__((ext_vector_type(4)));
typedef int v8i __attribute__((ext_vector_type(8)));
typedef int v4i __attribute__((ext_vector_type(4)));

#define N_TOT   8192
#define CDIM    128
#define HW_DIM  1024
#define TINV    10.0f     // 1 / TEMPERATURE
#define NCLS    8         // labels are 0..4; round up to 8
#define EPSV    1e-8f

#define JCHUNK_ROWS 64    // t-rows (columns of logits) staged per chunk
#define NCHUNKS     16    // 1024 columns per block / 64

// workspace layout (bytes)
#define OFF_SN16   (0u)                 // 8192*128 bf16  = 2 MB (normalized student, row-major)
#define OFF_TN16   (2097152u)           // 8192*128 bf16  = 2 MB (normalized teacher, row-major)
#define OFF_SF32   (4194304u)           // 8192*128 f32   = 4 MB (normalized student, fp32)
#define OFF_STATS  (8388608u)           // ClassSum[NCLS*128] f32 | hist[NCLS] i32 | rowSum[8192] f32 | acc[2] f32
#define STATS_WORDS (NCLS*CDIM + NCLS + N_TOT + 2)

#if defined(__has_builtin)
#if __has_builtin(__builtin_amdgcn_tensor_load_to_lds) && __has_builtin(__builtin_amdgcn_s_wait_tensorcnt)
#define HAVE_TDM 1
#endif
#endif
#ifndef HAVE_TDM
#define HAVE_TDM 0
#endif
#if defined(__has_include)
#if __has_include(<hip/amd_detail/amd_gfx1250_TDM.h>)
#define TDM_ARITY 6
#endif
#endif
#ifndef TDM_ARITY
#define TDM_ARITY 5
#endif

union Frag { v16bf v; u32x4 q[2]; };

// ---------------- K0: zero the reduction state ----------------
__global__ void k_zero(char* __restrict__ ws) {
    float* p = (float*)(ws + OFF_STATS);
    for (int i = blockIdx.x * blockDim.x + threadIdx.x; i < STATS_WORDS;
         i += gridDim.x * blockDim.x)
        p[i] = 0.0f;
}

// ---------------- K1: transpose [B,C,HW]->[N,C], L2-normalize, pack bf16,
//                  and accumulate per-class teacher sums + class histogram ----
__global__ void k_normalize(const float* __restrict__ s,
                            const float* __restrict__ t,
                            const int* __restrict__ tl,
                            char* __restrict__ ws) {
    const int n   = blockIdx.x;        // row 0..8191
    const int isT = blockIdx.y;        // 0 = student, 1 = teacher
    const int c   = threadIdx.x;       // 0..127 (channel)
    const int b   = n >> 10;
    const int hw  = n & (HW_DIM - 1);
    const float* src = isT ? t : s;
    float x = src[(size_t)b * CDIM * HW_DIM + (size_t)c * HW_DIM + hw];

    float ss = x * x;
    #pragma unroll
    for (int m = 16; m; m >>= 1) ss += __shfl_xor(ss, m, 32);
    __shared__ float red[4];
    if ((threadIdx.x & 31) == 0) red[threadIdx.x >> 5] = ss;
    __syncthreads();
    float tot = red[0] + red[1] + red[2] + red[3];
    float xn = x * rsqrtf(tot);

    bf16_t* q16 = (bf16_t*)(ws + (isT ? OFF_TN16 : OFF_SN16));
    q16[(size_t)n * CDIM + c] = (bf16_t)xn;

    if (!isT) {
        float* qf = (float*)(ws + OFF_SF32);
        qf[(size_t)n * CDIM + c] = xn;
    } else {
        int lbl = tl[n] & (NCLS - 1);
        float* cls = (float*)(ws + OFF_STATS);
        atomicAdd(&cls[lbl * CDIM + c], xn);
        if (c == 0) {
            int* hist = (int*)(ws + OFF_STATS + NCLS * CDIM * 4);
            atomicAdd(&hist[lbl], 1);
        }
    }
}

#if HAVE_TDM
// Issue one TDM 2D tile load: 64 rows x 256B (64 dwords) from global -> LDS.
__device__ __forceinline__ void tdm_issue(const bf16_t* gsrc, unsigned lds_byte_off) {
    unsigned long long ga = (unsigned long long)(uintptr_t)gsrc;
    // D# group0: count=1 | lds_addr | global_addr[56:0] | type=2
    v4u g0 = { 1u,
               lds_byte_off,
               (unsigned)(ga & 0xFFFFFFFFu),
               (unsigned)((ga >> 32) & 0x01FFFFFFu) | (2u << 30) };
    // D# group1: data_size=4B (code 2); tensor_dim0=64, tensor_dim1=64,
    // tile_dim0=64, tile_dim1=64, tensor_dim0_stride=64 dwords (=256B row)
    v8i g1 = { (int)(2u << 16),      // wg_mask=0, data_size=2, no flags
               (int)(64u << 16),     // tensor_dim0[15:0] << 16
               (int)(64u << 16),     // tensor_dim0 hi=0 | tensor_dim1[15:0]<<16
               (int)(64u << 16),     // tensor_dim1 hi=0 | tile_dim0<<16
               64,                   // tile_dim1 (tile_dim2=0)
               64,                   // tensor_dim0_stride lo
               0, 0 };
    v4i gz = { 0, 0, 0, 0 };
#if TDM_ARITY == 6
    v8i gz8 = { 0, 0, 0, 0, 0, 0, 0, 0 };
    __builtin_amdgcn_tensor_load_to_lds(g0, g1, gz, gz, gz8, 0);
#else
    __builtin_amdgcn_tensor_load_to_lds(g0, g1, gz, gz, 0);
#endif
}
#endif

// ---------------- K2: streamed logits + row sum-exp via bf16 WMMA -----------
// grid (64, 8), block 256 (8 waves). Wave w owns rows itile..itile+15 (A strip
// held in 4 x v16bf registers). The 1024-column B panel is staged through LDS
// in 16 double-buffered 64-column chunks by the Tensor Data Mover, shared by
// all 8 waves (8x L2 traffic reduction). logits in [-10,10] -> fixed-shift
// sum-exp is fp32 safe (no online max needed).
__global__ void __launch_bounds__(256, 1) k_logits_lse(char* __restrict__ ws) {
    const int lane = threadIdx.x & 31;
    const int wave = threadIdx.x >> 5;
    const int m    = lane & 15;        // row (A) / col (B) within 16x16 tile
    const int hi   = lane >> 4;        // lane half selects K sub-ranges
    const int itile = blockIdx.x * 128 + wave * 16;

    const bf16_t* Sn = (const bf16_t*)(ws + OFF_SN16);
    const bf16_t* Tn = (const bf16_t*)(ws + OFF_TN16);

    __shared__ __align__(16) bf16_t smem[2][JCHUNK_ROWS * CDIM];   // 2 x 16 KB

    // A fragments: 16x32 bf16; lane half hi: K = kb..kb+7 and kb+16..kb+23
    Frag a[4];
    const bf16_t* arow = Sn + (size_t)(itile + m) * CDIM;
    #pragma unroll
    for (int k4 = 0; k4 < 4; ++k4) {
        const int kb = k4 * 32 + hi * 8;
        a[k4].q[0] = *(const u32x4*)(arow + kb);
        a[k4].q[1] = *(const u32x4*)(arow + kb + 16);
    }

    float sums[8] = {0, 0, 0, 0, 0, 0, 0, 0};
    const int j0 = blockIdx.y * 1024;

#if HAVE_TDM
    const unsigned ldsBase = (unsigned)(uintptr_t)(void*)&smem[0][0];
    const unsigned bufStride = JCHUNK_ROWS * CDIM * 2u;   // bytes per buffer

    if (wave == 0)
        tdm_issue(Tn + (size_t)j0 * CDIM, ldsBase);
    if (wave == 0)
        __builtin_amdgcn_s_wait_tensorcnt(0);
    __syncthreads();

    for (int c = 0; c < NCHUNKS; ++c) {
        if (wave == 0 && (c + 1) < NCHUNKS)
            tdm_issue(Tn + (size_t)(j0 + (c + 1) * JCHUNK_ROWS) * CDIM,
                      ldsBase + ((unsigned)(c + 1) & 1u) * bufStride);

        const bf16_t* bbase = &smem[c & 1][0];
        #pragma unroll
        for (int jl = 0; jl < 4; ++jl) {
            const bf16_t* brow = bbase + (jl * 16 + m) * CDIM + hi * 8;
            v8f cc = {0, 0, 0, 0, 0, 0, 0, 0};
            #pragma unroll
            for (int k4 = 0; k4 < 4; ++k4) {
                Frag bf;
                bf.q[0] = *(const u32x4*)(brow + k4 * 32);
                bf.q[1] = *(const u32x4*)(brow + k4 * 32 + 16);
                cc = __builtin_amdgcn_wmma_f32_16x16x32_bf16(
                         false, a[k4].v, false, bf.v, (short)0, cc, false, false);
            }
            #pragma unroll
            for (int r = 0; r < 8; ++r) sums[r] += __expf(cc[r] * TINV);
        }

        if (wave == 0)
            __builtin_amdgcn_s_wait_tensorcnt(0);
        asm volatile("" ::: "memory");
        __syncthreads();
    }
#else
    // Fallback: cooperative global->LDS staging (single buffer).
    for (int c = 0; c < NCHUNKS; ++c) {
        __syncthreads();
        {   // 16 KB chunk: 1024 dwordx4; 256 threads x 4 each
            const u32x4* src = (const u32x4*)(Tn + (size_t)(j0 + c * JCHUNK_ROWS) * CDIM);
            u32x4* dst = (u32x4*)&smem[0][0];
            #pragma unroll
            for (int u = 0; u < 4; ++u)
                dst[threadIdx.x * 4 + u] = src[threadIdx.x * 4 + u];
        }
        __syncthreads();

        const bf16_t* bbase = &smem[0][0];
        #pragma unroll
        for (int jl = 0; jl < 4; ++jl) {
            const bf16_t* brow = bbase + (jl * 16 + m) * CDIM + hi * 8;
            v8f cc = {0, 0, 0, 0, 0, 0, 0, 0};
            #pragma unroll
            for (int k4 = 0; k4 < 4; ++k4) {
                Frag bf;
                bf.q[0] = *(const u32x4*)(brow + k4 * 32);
                bf.q[1] = *(const u32x4*)(brow + k4 * 32 + 16);
                cc = __builtin_amdgcn_wmma_f32_16x16x32_bf16(
                         false, a[k4].v, false, bf.v, (short)0, cc, false, false);
            }
            #pragma unroll
            for (int r = 0; r < 8; ++r) sums[r] += __expf(cc[r] * TINV);
        }
    }
#endif

    // reduce the 16 column-lanes within each half-wave
    #pragma unroll
    for (int msk = 1; msk < 16; msk <<= 1)
        #pragma unroll
        for (int r = 0; r < 8; ++r) sums[r] += __shfl_xor(sums[r], msk, 16);

    if (m == 0) {
        float* rowSum = (float*)(ws + OFF_STATS + (NCLS * CDIM + NCLS) * 4);
        #pragma unroll
        for (int r = 0; r < 8; ++r)
            atomicAdd(&rowSum[itile + hi * 8 + r], sums[r]);
    }
}

// ---------------- K3: per-row numerator via class sums + loss reduction -----
__global__ void k_finalize(const int* __restrict__ sl, char* __restrict__ ws) {
    const int lane = threadIdx.x & 31;
    const int wave = threadIdx.x >> 5;
    const int i = blockIdx.x * 8 + wave;            // one wave per anchor row

    const float* Sf     = (const float*)(ws + OFF_SF32);
    const float* cls    = (const float*)(ws + OFF_STATS);
    const int*   hist   = (const int*)  (ws + OFF_STATS + NCLS * CDIM * 4);
    const float* rowSum = (const float*)(ws + OFF_STATS + (NCLS * CDIM + NCLS) * 4);
    float*       acc    = (float*)      (ws + OFF_STATS + (NCLS * CDIM + NCLS + N_TOT) * 4);

    const int lbl = sl[i] & (NCLS - 1);
    float p = 0.0f;
    #pragma unroll
    for (int u = 0; u < 4; ++u) {
        const int k = lane * 4 + u;
        p += Sf[(size_t)i * CDIM + k] * cls[lbl * CDIM + k];
    }
    #pragma unroll
    for (int msk = 16; msk; msk >>= 1) p += __shfl_xor(p, msk, 32);

    if (lane == 0) {
        const float cnt = (float)hist[lbl];
        if (cnt > 0.5f && lbl != 0) {               // valid anchor & non-bg
            const float lse  = __logf(rowSum[i]);
            const float pos  = p * TINV;            // sum of matched logits
            const float mean = (pos - cnt * lse) / (cnt + EPSV);
            atomicAdd(&acc[0], -mean);
            atomicAdd(&acc[1], 1.0f);
        }
    }
}

// ---------------- K4: final scalar ----------------
__global__ void k_out(const char* __restrict__ ws, float* __restrict__ out) {
    const float* acc = (const float*)(ws + OFF_STATS + (NCLS * CDIM + NCLS + N_TOT) * 4);
    out[0] = acc[0] / acc[1];
}

extern "C" void kernel_launch(void* const* d_in, const int* in_sizes, int n_in,
                              void* d_out, int out_size, void* d_ws, size_t ws_size,
                              hipStream_t stream) {
    const float* s  = (const float*)d_in[0];
    const float* t  = (const float*)d_in[1];
    const int*   sl = (const int*)d_in[2];
    const int*   tl = (const int*)d_in[3];
    char*  ws  = (char*)d_ws;
    float* out = (float*)d_out;

    k_zero      <<<dim3((STATS_WORDS + 255) / 256), 256, 0, stream>>>(ws);
    k_normalize <<<dim3(N_TOT, 2),                  128, 0, stream>>>(s, t, tl, ws);
    k_logits_lse<<<dim3(64, 8),                     256, 0, stream>>>(ws);
    k_finalize  <<<dim3(N_TOT / 8),                 256, 0, stream>>>(sl, ws);
    k_out       <<<1, 1, 0, stream>>>(ws, out);
}
